// EditDistNeuralModelConcurrent_22273700397673
// MI455X (gfx1250) — compile-verified, matrix operand order checked by reading, not run on hardware
//
#include <hip/hip_runtime.h>
#include <hip/hip_bf16.h>
#include <math.h>

// ---------------------------------------------------------------------------
// Problem constants (from reference)
// ---------------------------------------------------------------------------
#define T_LEN   320
#define V_LEN   320
#define DMODEL  32
#define NHEAD   4
#define DHEAD   8
#define DFF     64
#define NLAYER  2
#define NCH     257          // C = 1 + 2*128
#define NCELL   (T_LEN * V_LEN)          // 102400
#define N_AS    ((size_t)NCELL * NCH)    // elements in a_s / ec

#define NEGF (-__builtin_inff())

typedef __attribute__((ext_vector_type(16))) _Float16 v16h;
typedef __attribute__((ext_vector_type(8)))  float    v8f;

// ---------------------------------------------------------------------------
// Helpers
// ---------------------------------------------------------------------------
__device__ __forceinline__ float lae(float a, float b) {
    float mx = fmaxf(a, b);
    if (mx == NEGF) return NEGF;
    float mn = fminf(a, b);
    return mx + log1pf(expf(mn - mx));
}
__device__ __forceinline__ float lae3(float a, float b, float c) {
    return lae(lae(a, b), c);
}

// K index inside a 16-bit WMMA A/B fragment (ISA 7.12.2):
// lane half hi=(lane>>4); element e of v16h -> K = e + 8*hi + (e>=8 ? 8 : 0)
__device__ __forceinline__ int wmma_k(int e, int hi) {
    return e + 8 * hi + ((e >= 8) ? 8 : 0);
}

__device__ __forceinline__ void layer_norm32(float* x, const float* g, const float* b) {
    float mu = 0.f;
    #pragma unroll
    for (int d = 0; d < 32; ++d) mu += x[d];
    mu *= (1.f / 32.f);
    float var = 0.f;
    #pragma unroll
    for (int d = 0; d < 32; ++d) { float t = x[d] - mu; var += t * t; }
    var *= (1.f / 32.f);
    float inv = rsqrtf(var + 1e-12f);
    #pragma unroll
    for (int d = 0; d < 32; ++d) x[d] = (x[d] - mu) * inv * g[d] + b[d];
}

// ---------------------------------------------------------------------------
// Kernel 1: tiny BERT encoder. grid.x selects sentence (0=AR, 1=EN).
// One thread per token; K/V shared via LDS; weights hit SMEM/L2 (uniform).
// ---------------------------------------------------------------------------
struct BertArgs {
    const float* p[2][37];   // leaves in insertion order (see launch)
    const int*   ids[2];
    float*       out[2];     // [320*32] encoded vectors
};

__global__ __launch_bounds__(T_LEN)
void bert_kernel(BertArgs args) {
    __shared__ float ksh[T_LEN][DMODEL];
    __shared__ float vsh[T_LEN][DMODEL];

    const int s = blockIdx.x;
    const int i = threadIdx.x;
    const float* const* P = args.p[s];

    const float* we = P[0];
    const float* pe = P[1];
    const float* te = P[2];

    int id = args.ids[s][i];
    float x[32];
    #pragma unroll
    for (int d = 0; d < 32; ++d) x[d] = we[id * 32 + d] + pe[i * 32 + d] + te[d];
    layer_norm32(x, P[3], P[4]);

    for (int l = 0; l < NLAYER; ++l) {
        const int b0 = 5 + 16 * l;
        const float* Wq = P[b0 + 0];  const float* bq = P[b0 + 1];
        const float* Wk = P[b0 + 2];  const float* bk = P[b0 + 3];
        const float* Wv = P[b0 + 4];  const float* bv = P[b0 + 5];
        const float* Wo = P[b0 + 6];  const float* bo = P[b0 + 7];
        const float* g1 = P[b0 + 8];  const float* bb1 = P[b0 + 9];
        const float* W1 = P[b0 + 10]; const float* c1 = P[b0 + 11];
        const float* W2 = P[b0 + 12]; const float* c2 = P[b0 + 13];
        const float* g2 = P[b0 + 14]; const float* bb2 = P[b0 + 15];

        float q[32];
        for (int n = 0; n < 32; ++n) {
            float aq = bq[n], ak = bk[n], av = bv[n];
            #pragma unroll
            for (int k = 0; k < 32; ++k) {
                float xv = x[k];
                aq += xv * Wq[k * 32 + n];
                ak += xv * Wk[k * 32 + n];
                av += xv * Wv[k * 32 + n];
            }
            q[n] = aq; ksh[i][n] = ak; vsh[i][n] = av;
        }
        __syncthreads();

        // Attention: online softmax per head (flash-style)
        float ctx[32];
        for (int h = 0; h < NHEAD; ++h) {
            float m = NEGF, ssum = 0.f;
            float acc[DHEAD];
            #pragma unroll
            for (int d = 0; d < DHEAD; ++d) acc[d] = 0.f;
            for (int j = 0; j < T_LEN; ++j) {
                float sc = 0.f;
                #pragma unroll
                for (int d = 0; d < DHEAD; ++d) sc += q[h * 8 + d] * ksh[j][h * 8 + d];
                sc *= 0.35355339059327373f;  // 1/sqrt(8)
                if (sc > m) {
                    float r = expf(m - sc);
                    ssum *= r;
                    #pragma unroll
                    for (int d = 0; d < DHEAD; ++d) acc[d] *= r;
                    m = sc;
                }
                float p = expf(sc - m);
                ssum += p;
                #pragma unroll
                for (int d = 0; d < DHEAD; ++d) acc[d] += p * vsh[j][h * 8 + d];
            }
            float inv = 1.f / ssum;
            #pragma unroll
            for (int d = 0; d < DHEAD; ++d) ctx[h * 8 + d] = acc[d] * inv;
        }

        // output projection + residual + LN
        float o[32];
        for (int n = 0; n < 32; ++n) {
            float a = bo[n];
            #pragma unroll
            for (int k = 0; k < 32; ++k) a += ctx[k] * Wo[k * 32 + n];
            o[n] = a;
        }
        #pragma unroll
        for (int d = 0; d < 32; ++d) x[d] += o[d];
        layer_norm32(x, g1, bb1);

        // FFN (exact gelu) + residual + LN
        float h1[DFF];
        for (int mm = 0; mm < DFF; ++mm) {
            float a = c1[mm];
            #pragma unroll
            for (int k = 0; k < 32; ++k) a += x[k] * W1[k * DFF + mm];
            h1[mm] = 0.5f * a * (1.f + erff(a * 0.7071067811865475f));
        }
        for (int n = 0; n < 32; ++n) {
            float a = c2[n];
            #pragma unroll
            for (int mm = 0; mm < DFF; ++mm) a += h1[mm] * W2[mm * 32 + n];
            x[n] += a;
        }
        layer_norm32(x, g2, bb2);
        __syncthreads();   // before ksh/vsh reuse next layer
    }

    float* outv = args.out[s];
    #pragma unroll
    for (int d = 0; d < 32; ++d) outv[i * 32 + d] = x[d];
}

// ---------------------------------------------------------------------------
// Kernel 2: action scores with WMMA. 128 threads = 4 waves; wave owns 16
// consecutive (t,v) cells.  GEMM1: [16x64]x[64x32] -> feat (ReLU); GEMM2:
// [16x32]x[32x257] -> logits; log-softmax over 257; write a_s + DP slices.
// ---------------------------------------------------------------------------
__global__ __launch_bounds__(128)
void scores_kernel(const float* __restrict__ arv, const float* __restrict__ env,
                   const float* __restrict__ Wp,  const float* __restrict__ bp,
                   const float* __restrict__ Wa,  const float* __restrict__ bac,
                   const int*   __restrict__ en_sent,
                   float* __restrict__ a_s,
                   float* __restrict__ s_del, float* __restrict__ s_ins,
                   float* __restrict__ s_sub,
                   float* __restrict__ g_ins2, float* __restrict__ g_sub2) {
    __shared__ _Float16 sWp[64 * 32];
    __shared__ _Float16 sWa[32 * NCH];
    __shared__ float    sBp[32];
    __shared__ float    sBa[NCH];
    __shared__ _Float16 sFeat[4][16][32];
    __shared__ float    sLogit[4][16][NCH];
    __shared__ float    sLse[4][16];

    const int tid  = threadIdx.x;
    const int w    = tid >> 5;
    const int lane = tid & 31;
    const int lo   = lane & 15;
    const int hi   = lane >> 4;

    for (int idx = tid; idx < 64 * 32; idx += 128) sWp[idx] = (_Float16)Wp[idx];
    for (int idx = tid; idx < 32 * NCH; idx += 128) sWa[idx] = (_Float16)Wa[idx];
    for (int idx = tid; idx < 32;  idx += 128) sBp[idx] = bp[idx];
    for (int idx = tid; idx < NCH; idx += 128) sBa[idx] = bac[idx];
    __syncthreads();

    const int base = (blockIdx.x * 4 + w) * 16;
    const int cell = base + lo;
    const int t = cell / V_LEN;
    const int v = cell - t * V_LEN;

    // ----- GEMM1 A fragments: K0..31 = ar_vec[t], K32..63 = en_vec[v]
    const float* fa = arv + t * 32;
    const float* fe = env + v * 32;
    v16h aA, aE;
    #pragma unroll
    for (int e = 0; e < 16; ++e) {
        int k = wmma_k(e, hi);
        aA[e] = (_Float16)fa[k];
        aE[e] = (_Float16)fe[k];
    }

    #pragma unroll
    for (int nt = 0; nt < 2; ++nt) {
        const int n = nt * 16 + lo;
        v16h b0, b1;
        #pragma unroll
        for (int e = 0; e < 16; ++e) {
            int k = wmma_k(e, hi);
            b0[e] = sWp[k * 32 + n];
            b1[e] = sWp[(32 + k) * 32 + n];
        }
        v8f acc = {};
        acc = __builtin_amdgcn_wmma_f32_16x16x32_f16(false, aA, false, b0, (short)0, acc, false, false);
        acc = __builtin_amdgcn_wmma_f32_16x16x32_f16(false, aE, false, b1, (short)0, acc, false, false);
        #pragma unroll
        for (int r = 0; r < 8; ++r) {
            int m = r + 8 * hi;
            float fv = fmaxf(acc[r] + sBp[n], 0.f);   // + bias, ReLU
            sFeat[w][m][n] = (_Float16)fv;
        }
    }
    __syncthreads();

    // ----- relayout feat C-fragment -> A-fragment via LDS
    v16h a2;
    #pragma unroll
    for (int e = 0; e < 16; ++e) a2[e] = sFeat[w][lo][wmma_k(e, hi)];

    // ----- GEMM2: 17 N-tiles over 257 channels
    for (int nt = 0; nt < 17; ++nt) {
        const int c = nt * 16 + lo;
        v16h b2;
        #pragma unroll
        for (int e = 0; e < 16; ++e) {
            int k = wmma_k(e, hi);
            b2[e] = (c < NCH) ? sWa[k * NCH + c] : (_Float16)0.f;
        }
        v8f acc = {};
        acc = __builtin_amdgcn_wmma_f32_16x16x32_f16(false, a2, false, b2, (short)0, acc, false, false);
        if (c < NCH) {
            #pragma unroll
            for (int r = 0; r < 8; ++r) sLogit[w][r + 8 * hi][c] = acc[r] + sBa[c];
        }
    }

    // ----- per-cell log-softmax (2 lanes / cell)
    {
        const int mycell = lane >> 1;
        const int par    = lane & 1;
        float mx = NEGF;
        for (int ch = par; ch < NCH; ch += 2) mx = fmaxf(mx, sLogit[w][mycell][ch]);
        mx = fmaxf(mx, __shfl_xor(mx, 1, 32));
        float sum = 0.f;
        for (int ch = par; ch < NCH; ch += 2) sum += expf(sLogit[w][mycell][ch] - mx);
        sum += __shfl_xor(sum, 1, 32);
        float lse = mx + logf(sum);
        if (par == 0) sLse[w][mycell] = lse;
    }

    // ----- coalesced a_s write (16 cells * 257 ch per wave)
    for (int idx = lane; idx < 16 * NCH; idx += 32) {
        int m  = idx / NCH;
        int ch = idx - m * NCH;
        a_s[(size_t)(base + m) * NCH + ch] = sLogit[w][m][ch] - sLse[w][m];
    }

    // ----- DP slices while logits are hot in LDS
    if (lane < 16) {
        const int cl = base + lane;
        const int tt = cl / V_LEN;
        const int vv = cl - tt * V_LEN;
        const float lse = sLse[w][lane];
        const int ech = en_sent[vv];
        s_del[cl] = sLogit[w][lane][0] - lse;
        s_ins[cl] = sLogit[w][lane][1 + ech] - lse;
        s_sub[cl] = sLogit[w][lane][129 + ech] - lse;
        if (vv >= 1) {
            const int ep = en_sent[vv - 1];
            g_ins2[cl] = sLogit[w][lane][1 + ep] - lse;    // a_s[t][v][ins_ch[v-1]]
            g_sub2[cl] = sLogit[w][lane][129 + ep] - lse;  // a_s[t][v][sub_ch[v-1]]
        } else {
            g_ins2[cl] = NEGF;
            g_sub2[cl] = NEGF;
        }
    }
}

// ---------------------------------------------------------------------------
// Kernel 3: forward alpha via anti-diagonal wavefront (639 steps, 1 block).
// ---------------------------------------------------------------------------
__global__ __launch_bounds__(T_LEN)
void alpha_kernel(const float* __restrict__ sd, const float* __restrict__ si,
                  const float* __restrict__ ss, float* __restrict__ alpha,
                  float* __restrict__ out_last) {
    __shared__ float dbuf[3][T_LEN];
    const int t = threadIdx.x;
    for (int d = 0; d < T_LEN + V_LEN - 1; ++d) {
        const int v = d - t;
        float up   = (t >= 1 && d >= 1)            ? dbuf[(d + 2) % 3][t - 1] : NEGF; // a[t-1][v]
        float left = (d >= 1 && v >= 1)            ? dbuf[(d + 2) % 3][t]     : NEGF; // a[t][v-1]
        float diag = (t >= 1 && d >= 2 && v >= 1)  ? dbuf[(d + 1) % 3][t - 1] : NEGF; // a[t-1][v-1]
        if (v >= 0 && v < V_LEN) {
            const int idx = t * V_LEN + v;
            float val;
            if (t == 0 && v == 0)      val = 0.f;
            else if (t == 0)           val = left + si[idx];
            else if (v == 0)           val = up + sd[idx];
            else                       val = lae3(up + sd[idx], diag + ss[idx], left + si[idx]);
            dbuf[d % 3][t] = val;
            alpha[idx] = val;
            if (t == T_LEN - 1 && v == V_LEN - 1) *out_last = val;
        }
        __syncthreads();
    }
}

// ---------------------------------------------------------------------------
// Kernel 4: backward beta via reversed wavefront.
// ---------------------------------------------------------------------------
__global__ __launch_bounds__(T_LEN)
void beta_kernel(const float* __restrict__ sd, const float* __restrict__ gi2,
                 const float* __restrict__ gs2, float* __restrict__ beta) {
    __shared__ float dbuf[3][T_LEN];
    const int t = threadIdx.x;
    for (int d = T_LEN + V_LEN - 2; d >= 0; --d) {
        const int v = d - t;
        float bdn = (t < T_LEN - 1 && d < 638)                          ? dbuf[(d + 1) % 3][t + 1] : NEGF; // b[t+1][v]
        float brt = (d < 638 && v >= 0 && v < V_LEN - 1)                ? dbuf[(d + 1) % 3][t]     : NEGF; // b[t][v+1]
        float bdg = (t < T_LEN - 1 && d < 637 && v >= 0 && v < V_LEN-1) ? dbuf[(d + 2) % 3][t + 1] : NEGF; // b[t+1][v+1]
        if (v >= 0 && v < V_LEN) {
            const int idx = t * V_LEN + v;
            const float init = (t == T_LEN - 1 && v == V_LEN - 1) ? 0.f : NEGF;
            const float bsd = (t < T_LEN - 1)                  ? sd[idx + V_LEN]        : NEGF;
            const float bss = (t < T_LEN - 1 && v < V_LEN - 1) ? gs2[idx + V_LEN + 1]   : NEGF;
            const float bsi = (v < V_LEN - 1)                  ? gi2[idx + 1]           : NEGF;
            float cc  = lae3(init, bsd + bdn, bss + bdg);
            float val = lae(cc, bsi + brt);
            dbuf[d % 3][t] = val;
            beta[idx] = val;
        }
        __syncthreads();
    }
}

// ---------------------------------------------------------------------------
// Kernel 5: expected counts.  One wave per cell; only channels
// {0, 1+en[v], 129+en[v]} are finite -> everything else is 0 (or NaN row).
// 105 MB pure-store pass.
// ---------------------------------------------------------------------------
__global__ __launch_bounds__(256)
void ec_kernel(const float* __restrict__ alpha, const float* __restrict__ beta,
               const float* __restrict__ sdl,   const float* __restrict__ sin_,
               const float* __restrict__ ssb,   const int* __restrict__ en,
               float* __restrict__ ec) {
    const int wid  = blockIdx.x * (blockDim.x >> 5) + (threadIdx.x >> 5);
    const int lane = threadIdx.x & 31;
    if (wid >= NCELL) return;
    const int t = wid / V_LEN;
    const int v = wid - t * V_LEN;
    const int idx = wid;

    __builtin_prefetch(&beta[idx], 0, 0);   // gfx1250 global_prefetch

    const float bt = beta[idx];
    const float vd = (t >= 1 && t <= T_LEN - 2) ? alpha[idx - V_LEN] + sdl[idx] + bt : NEGF;
    const float vi = (v >= 1 && v <= V_LEN - 2) ? alpha[idx - 1] + sin_[idx] + bt : NEGF;
    const float vs = (t >= 1 && t <= T_LEN - 2 && v >= 1 && v <= V_LEN - 2)
                         ? alpha[idx - V_LEN - 1] + ssb[idx] + bt : NEGF;

    const float m = fmaxf(vd, fmaxf(vi, vs));
    float pd, pi, ps;
    bool nanrow = false;
    if (m == NEGF) {
        nanrow = true;
        pd = pi = ps = __builtin_nanf("");
    } else {
        const float lse = m + logf(expf(vd - m) + expf(vi - m) + expf(vs - m));
        pd = expf(vd - lse);
        pi = expf(vi - lse);
        ps = expf(vs - lse);
    }
    const int ci = 1 + en[v];
    const int cs = 129 + en[v];
    const float nanv = __builtin_nanf("");
    float* row = ec + (size_t)idx * NCH;
    for (int ch = lane; ch < NCH; ch += 32) {
        float o;
        if (nanrow) o = nanv;
        else        o = (ch == 0) ? pd : (ch == ci) ? pi : (ch == cs) ? ps : 0.f;
        row[ch] = o;
    }
}

// ---------------------------------------------------------------------------
// Launch
// ---------------------------------------------------------------------------
extern "C" void kernel_launch(void* const* d_in, const int* in_sizes, int n_in,
                              void* d_out, int out_size, void* d_ws, size_t ws_size,
                              hipStream_t stream) {
    (void)in_sizes; (void)n_in; (void)out_size; (void)ws_size;

    // setup_inputs() insertion order:
    // 0: ar_sent, 1: en_sent, 2..38: ar_params (37 leaves), 39..75: en_params,
    // 76: Wp, 77: bp, 78: Wa, 79: ba
    const int* ar_sent = (const int*)d_in[0];
    const int* en_sent = (const int*)d_in[1];

    BertArgs bargs;
    for (int j = 0; j < 37; ++j) {
        bargs.p[0][j] = (const float*)d_in[2 + j];
        bargs.p[1][j] = (const float*)d_in[39 + j];
    }
    bargs.ids[0] = ar_sent;
    bargs.ids[1] = en_sent;

    const float* Wp  = (const float*)d_in[76];
    const float* bp  = (const float*)d_in[77];
    const float* Wa  = (const float*)d_in[78];
    const float* bav = (const float*)d_in[79];

    // workspace layout (floats)
    float* ws     = (float*)d_ws;
    float* arv    = ws;                       // 320*32
    float* env    = arv + T_LEN * DMODEL;     // 320*32
    float* s_del  = env + V_LEN * DMODEL;     // 320*320 each below
    float* s_ins  = s_del  + NCELL;
    float* s_sub  = s_ins  + NCELL;
    float* g_ins2 = s_sub  + NCELL;
    float* g_sub2 = g_ins2 + NCELL;
    float* alpha  = g_sub2 + NCELL;
    float* beta   = alpha  + NCELL;

    bargs.out[0] = arv;
    bargs.out[1] = env;

    float* out      = (float*)d_out;
    float* a_s      = out;                    // (T,V,C)
    float* ec       = out + N_AS;             // (T,V,C)
    float* out_last = out + 2 * N_AS;         // alpha[-1,-1]

    bert_kernel<<<2, T_LEN, 0, stream>>>(bargs);
    scores_kernel<<<NCELL / 64, 128, 0, stream>>>(arv, env, Wp, bp, Wa, bav, en_sent,
                                                  a_s, s_del, s_ins, s_sub, g_ins2, g_sub2);
    alpha_kernel<<<1, T_LEN, 0, stream>>>(s_del, s_ins, s_sub, alpha, out_last);
    beta_kernel<<<1, T_LEN, 0, stream>>>(s_del, g_ins2, g_sub2, beta);
    ec_kernel<<<NCELL / 8, 256, 0, stream>>>(alpha, beta, s_del, s_ins, s_sub, en_sent, ec);
}
